// SparseResBlock3D_14310831031003
// MI455X (gfx1250) — matrix operand order repaired and checked by reading, not compiled, not run on hardware
//
#include <hip/hip_runtime.h>

#define NPTS  100000
#define DIMC  256
#define INTER 64
#define KOFF  27
#define EPSV  1e-5f
#define TILES (NPTS / 16)          // 6250 exactly

typedef __bf16 bf16_t;
typedef __bf16          bf16x16 __attribute__((ext_vector_type(16)));
typedef float           v8f     __attribute__((ext_vector_type(8)));
typedef unsigned short  us8     __attribute__((ext_vector_type(8)));

union Frag {
    bf16_t  e[16];
    us8     h[2];
    bf16x16 v;
};

// ---- fragment loaders (wave32 WMMA layouts, ISA 7.12.2) -------------------
// A 16x32 bf16: lane (half,r): rows M=r, K = {k0+half*8 .. +7, k0+16+half*8 .. +7}
__device__ __forceinline__ bf16x16 load_a_bf16(const bf16_t* rowptr, int k0, int half) {
    Frag f;
    f.h[0] = *(const us8*)(rowptr + k0 + half * 8);
    f.h[1] = *(const us8*)(rowptr + k0 + 16 + half * 8);
    return f.v;
}
// A 16x32 from f32 source: native f32->bf16 converts (packed cvt on gfx1250)
__device__ __forceinline__ bf16x16 load_a_f32(const float* rowptr, int k0, int half) {
    Frag f;
#pragma unroll
    for (int j = 0; j < 8; ++j) f.e[j]     = (bf16_t)rowptr[k0 + half * 8 + j];
#pragma unroll
    for (int j = 0; j < 8; ++j) f.e[8 + j] = (bf16_t)rowptr[k0 + 16 + half * 8 + j];
    return f.v;
}
// B 32x16 bf16 from transposed weights wt[Ncol][ldk]: lane (half,r): col N=n0+r,
// K = {k0+half*8 .. +7, k0+16+half*8 .. +7} (same order as A)
__device__ __forceinline__ bf16x16 load_b(const bf16_t* wt, int ldk,
                                          int n0, int k0, int r, int half) {
    Frag f;
    const bf16_t* p = wt + (size_t)(n0 + r) * ldk + k0 + half * 8;
    f.h[0] = *(const us8*)p;
    f.h[1] = *(const us8*)(p + 16);
    return f.v;
}

#define WMMA_BF16(A, B, C) \
    __builtin_amdgcn_wmma_f32_16x16x32_bf16(false, (A), false, (B), (short)0, (C), false, false)

__device__ __forceinline__ void wave_reduce_and_atomic(float s, float sq, int lane,
                                                       double* dsum, double* dsq) {
#pragma unroll
    for (int off = 16; off >= 1; off >>= 1) {
        s  += __shfl_xor(s,  off, 32);
        sq += __shfl_xor(sq, off, 32);
    }
    if (lane == 0) {
        atomicAdd(dsum, (double)s);
        atomicAdd(dsq,  (double)sq);
    }
}

// ---- kernel 0: transpose weights to bf16 [Ncol][K] + zero stats -----------
__global__ void prep_kernel(const float* __restrict__ W1, const float* __restrict__ W2,
                            const float* __restrict__ W3,
                            bf16_t* __restrict__ w1t, bf16_t* __restrict__ w2t,
                            bf16_t* __restrict__ w3t, double* __restrict__ stats) {
    int i = blockIdx.x * blockDim.x + threadIdx.x;
    if (i < 8) stats[i] = 0.0;
    if (i < INTER * DIMC) {                    // W1T [64][256] <- W1 [256][64]
        int d = i >> 8, c = i & 255;
        w1t[i] = (bf16_t)W1[c * INTER + d];
    }
    if (i < DIMC * INTER) {                    // W3T [256][64] <- W3 [64][256]
        int d = i >> 6, c = i & 63;
        w3t[i] = (bf16_t)W3[c * DIMC + d];
    }
    if (i < KOFF * INTER * INTER) {            // W2T [27][64][64] <- W2 [27][64][64] (c<->d)
        int k = i >> 12, rem = i & 4095;
        int d = rem >> 6, c = rem & 63;
        w2t[i] = (bf16_t)W2[k * 4096 + c * INTER + d];
    }
}

// ---- kernel 1: stats of x @ W1 --------------------------------------------
__global__ void conv1_stats(const float* __restrict__ x, const bf16_t* __restrict__ w1t,
                            double* __restrict__ stats) {
    int w = blockIdx.x * 4 + (threadIdx.x >> 5);
    if (w >= TILES) return;
    int lane = threadIdx.x & 31, r = lane & 15, half = lane >> 4;
    const float* rowptr = x + (size_t)(w * 16 + r) * DIMC;
    v8f acc[4] = {};
    for (int k0 = 0; k0 < DIMC; k0 += 32) {
        bf16x16 a = load_a_f32(rowptr, k0, half);
#pragma unroll
        for (int nt = 0; nt < 4; ++nt) {
            bf16x16 b = load_b(w1t, DIMC, nt * 16, k0, r, half);
            acc[nt] = WMMA_BF16(a, b, acc[nt]);
        }
    }
    float s = 0.f, sq = 0.f;
#pragma unroll
    for (int nt = 0; nt < 4; ++nt)
#pragma unroll
        for (int g = 0; g < 8; ++g) { float v = acc[nt][g]; s += v; sq += v * v; }
    wave_reduce_and_atomic(s, sq, lane, &stats[0], &stats[1]);
}

// ---- kernel 2: recompute x @ W1, GN1 + ReLU -> h1n (bf16) -----------------
__global__ void conv1_norm(const float* __restrict__ x, const bf16_t* __restrict__ w1t,
                           const float* __restrict__ g1, const float* __restrict__ b1,
                           const double* __restrict__ stats, bf16_t* __restrict__ h1n) {
    int w = blockIdx.x * 4 + (threadIdx.x >> 5);
    if (w >= TILES) return;
    int lane = threadIdx.x & 31, r = lane & 15, half = lane >> 4;
    const float* rowptr = x + (size_t)(w * 16 + r) * DIMC;
    v8f acc[4] = {};
    for (int k0 = 0; k0 < DIMC; k0 += 32) {
        bf16x16 a = load_a_f32(rowptr, k0, half);
#pragma unroll
        for (int nt = 0; nt < 4; ++nt) {
            bf16x16 b = load_b(w1t, DIMC, nt * 16, k0, r, half);
            acc[nt] = WMMA_BF16(a, b, acc[nt]);
        }
    }
    double cnt = (double)NPTS * INTER;
    double mu  = stats[0] / cnt;
    double var = stats[1] / cnt - mu * mu;
    float rs   = rsqrtf((float)var + EPSV);
#pragma unroll
    for (int nt = 0; nt < 4; ++nt) {
        int c   = nt * 16 + r;
        float A = g1[c] * rs;
        float B = b1[c] - (float)mu * A;
#pragma unroll
        for (int g = 0; g < 8; ++g) {
            int row = w * 16 + half * 8 + g;
            float v = fmaxf(acc[nt][g] * A + B, 0.f);
            h1n[(size_t)row * INTER + c] = (bf16_t)v;
        }
    }
}

// ---- kernel 3: conv2 gather-GEMM -> h2raw (f32) + stats -------------------
__global__ void conv2_kernel(const int* __restrict__ nidx, const bf16_t* __restrict__ h1n,
                             const bf16_t* __restrict__ w2t,
                             float* __restrict__ h2raw, double* __restrict__ stats) {
    int w = blockIdx.x * 4 + (threadIdx.x >> 5);
    if (w >= TILES) return;
    int lane = threadIdx.x & 31, r = lane & 15, half = lane >> 4;
    int p = w * 16 + r;
    v8f acc[4] = {};
    for (int k = 0; k < KOFF; ++k) {
        int nb = nidx[p * KOFF + k];
        const bf16_t* rowptr = h1n + (size_t)nb * INTER;
        if (k + 1 < KOFF) {   // prefetch next gathered row (global_prefetch_b8)
            int nb2 = nidx[p * KOFF + k + 1];
            __builtin_prefetch(h1n + (size_t)nb2 * INTER, 0, 0);
        }
        const bf16_t* wt = w2t + k * (INTER * INTER);
#pragma unroll
        for (int kc = 0; kc < INTER; kc += 32) {
            bf16x16 a = load_a_bf16(rowptr, kc, half);
#pragma unroll
            for (int nt = 0; nt < 4; ++nt) {
                bf16x16 b = load_b(wt, INTER, nt * 16, kc, r, half);
                acc[nt] = WMMA_BF16(a, b, acc[nt]);
            }
        }
    }
    float s = 0.f, sq = 0.f;
#pragma unroll
    for (int nt = 0; nt < 4; ++nt)
#pragma unroll
        for (int g = 0; g < 8; ++g) {
            float v = acc[nt][g];
            h2raw[(size_t)(w * 16 + half * 8 + g) * INTER + nt * 16 + r] = v;
            s += v; sq += v * v;
        }
    wave_reduce_and_atomic(s, sq, lane, &stats[2], &stats[3]);
}

// ---- kernel 4: GN2 + ReLU elementwise -> h2n (bf16) -----------------------
__global__ void gn2_kernel(const float* __restrict__ h2raw, const float* __restrict__ g2,
                           const float* __restrict__ b2, const double* __restrict__ stats,
                           bf16_t* __restrict__ h2n) {
    double cnt = (double)NPTS * INTER;
    double mu  = stats[2] / cnt;
    double var = stats[3] / cnt - mu * mu;
    float rs   = rsqrtf((float)var + EPSV);
    size_t base = ((size_t)blockIdx.x * blockDim.x + threadIdx.x) * 4;
#pragma unroll
    for (int j = 0; j < 4; ++j) {
        size_t i = base + j;
        int c   = (int)(i & 63);
        float A = g2[c] * rs;
        float B = b2[c] - (float)mu * A;
        h2n[i] = (bf16_t)fmaxf(h2raw[i] * A + B, 0.f);
    }
}

// ---- kernel 5: stats of h2n @ W3 ------------------------------------------
__global__ void conv3_stats(const bf16_t* __restrict__ h2n,
                            const bf16_t* __restrict__ w3t, double* __restrict__ stats) {
    int w = blockIdx.x * 4 + (threadIdx.x >> 5);   // 25000 waves: tile x col-quad
    int t = w >> 2, q = w & 3;
    if (t >= TILES) return;
    int lane = threadIdx.x & 31, r = lane & 15, half = lane >> 4;
    const bf16_t* rowptr = h2n + (size_t)(t * 16 + r) * INTER;
    v8f acc[4] = {};
#pragma unroll
    for (int kc = 0; kc < INTER; kc += 32) {
        bf16x16 a = load_a_bf16(rowptr, kc, half);
#pragma unroll
        for (int nt = 0; nt < 4; ++nt) {
            bf16x16 b = load_b(w3t, INTER, q * 64 + nt * 16, kc, r, half);
            acc[nt] = WMMA_BF16(a, b, acc[nt]);
        }
    }
    float s = 0.f, sq = 0.f;
#pragma unroll
    for (int nt = 0; nt < 4; ++nt)
#pragma unroll
        for (int g = 0; g < 8; ++g) { float v = acc[nt][g]; s += v; sq += v * v; }
    wave_reduce_and_atomic(s, sq, lane, &stats[4], &stats[5]);
}

// ---- kernel 6: recompute conv3, GN3, +x residual, ReLU -> out -------------
__global__ void final_kernel(const float* __restrict__ x, const bf16_t* __restrict__ h2n,
                             const bf16_t* __restrict__ w3t,
                             const float* __restrict__ g3, const float* __restrict__ b3,
                             const double* __restrict__ stats, float* __restrict__ out) {
    int w = blockIdx.x * 4 + (threadIdx.x >> 5);
    int t = w >> 2, q = w & 3;
    if (t >= TILES) return;
    int lane = threadIdx.x & 31, r = lane & 15, half = lane >> 4;
    const bf16_t* rowptr = h2n + (size_t)(t * 16 + r) * INTER;
    v8f acc[4] = {};
#pragma unroll
    for (int kc = 0; kc < INTER; kc += 32) {
        bf16x16 a = load_a_bf16(rowptr, kc, half);
#pragma unroll
        for (int nt = 0; nt < 4; ++nt) {
            bf16x16 b = load_b(w3t, INTER, q * 64 + nt * 16, kc, r, half);
            acc[nt] = WMMA_BF16(a, b, acc[nt]);
        }
    }
    double cnt = (double)NPTS * DIMC;
    double mu  = stats[4] / cnt;
    double var = stats[5] / cnt - mu * mu;
    float rs   = rsqrtf((float)var + EPSV);
#pragma unroll
    for (int nt = 0; nt < 4; ++nt) {
        int c   = q * 64 + nt * 16 + r;
        float A = g3[c] * rs;
        float B = b3[c] - (float)mu * A;
#pragma unroll
        for (int g = 0; g < 8; ++g) {
            int row  = t * 16 + half * 8 + g;
            size_t o = (size_t)row * DIMC + c;
            out[o]   = fmaxf(x[o] + acc[nt][g] * A + B, 0.f);
        }
    }
}

extern "C" void kernel_launch(void* const* d_in, const int* in_sizes, int n_in,
                              void* d_out, int out_size, void* d_ws, size_t ws_size,
                              hipStream_t stream) {
    const float* x    = (const float*)d_in[0];
    const int*   nidx = (const int*)  d_in[1];
    const float* W1   = (const float*)d_in[2];
    const float* g1   = (const float*)d_in[3];
    const float* b1   = (const float*)d_in[4];
    const float* W2   = (const float*)d_in[5];
    const float* g2   = (const float*)d_in[6];
    const float* b2   = (const float*)d_in[7];
    const float* W3   = (const float*)d_in[8];
    const float* g3   = (const float*)d_in[9];
    const float* b3   = (const float*)d_in[10];
    float* out = (float*)d_out;

    char* ws = (char*)d_ws;
    bf16_t* w1t   = (bf16_t*)(ws + 0);          //  32768 B  [64][256]
    bf16_t* w3t   = (bf16_t*)(ws + 32768);      //  32768 B  [256][64]
    bf16_t* w2t   = (bf16_t*)(ws + 65536);      // 221184 B  [27][64][64]
    double* stats = (double*)(ws + 286720);     //     64 B
    bf16_t* h1n   = (bf16_t*)(ws + 286784);     // 12.8 MB  bf16 [N][64]
    float*  h2raw = (float*) (ws + 13086784);   // 25.6 MB  f32  [N][64]
    bf16_t* h2n   = (bf16_t*)(ws + 38686784);   // 12.8 MB  bf16 [N][64]
    // total workspace usage: 51,486,784 bytes

    prep_kernel  <<<432,  256, 0, stream>>>(W1, W2, W3, w1t, w2t, w3t, stats);
    conv1_stats  <<<1563, 128, 0, stream>>>(x, w1t, stats);
    conv1_norm   <<<1563, 128, 0, stream>>>(x, w1t, g1, b1, stats, h1n);
    conv2_kernel <<<1563, 128, 0, stream>>>(nidx, h1n, w2t, h2raw, stats);
    gn2_kernel   <<<6250, 256, 0, stream>>>(h2raw, g2, b2, stats, h2n);
    conv3_stats  <<<6250, 128, 0, stream>>>(h2n, w3t, stats);
    final_kernel <<<6250, 128, 0, stream>>>(x, h2n, w3t, g3, b3, stats, out);
}
